// MultiHeadAttentionLayer_63299228008901
// MI455X (gfx1250) — matrix operand order repaired and checked by reading, not compiled, not run on hardware
//
#include <hip/hip_runtime.h>
#include <math.h>

// ---------------------------------------------------------------------------
// MHA forward for MI455X (gfx1250, wave32, WMMA bf16 16x16x32, f32 accum).
//
// Pipeline:
//   0) convert f32 activations/weights -> bf16 (NT loads; bf16 set stays in L2)
//   1) qkv_proj_kernel: X@W^T+bias, fused RoPE for Q/K; Q,K -> [B,H,S,HD],
//      V -> [B,H,HD,S] (transposed so attn@V B-operand is contiguous)
//   2) softmax_stats_kernel: online (max,sum) of causal scaled scores per row
//   3) attn_ctx_kernel: recompute scores, write normalized attn with
//      NON-TEMPORAL stores (537MB streaming, don't evict L2-resident GEMM
//      operands), fuse P@V via LDS transpose of P into A-layout
//   4) oproj_kernel: ctx@Wo^T + bo -> f32 out (NT stores)
// ---------------------------------------------------------------------------

typedef __attribute__((ext_vector_type(8)))  __bf16 v8bf;
typedef __attribute__((ext_vector_type(16))) __bf16 v16bf;
typedef __attribute__((ext_vector_type(8)))  float  v8f;
typedef __attribute__((ext_vector_type(4)))  float  f4v;

#define DM   1024
#define SEQ  2048
#define NB   2
#define NH   16
#define HDIM 64
#define NTOK (NB * SEQ)   // 4096

__device__ __forceinline__ v8f wmma_bf16(v16bf a, v16bf b, v8f c) {
  // D = A(16x32) * B(32x16) + C, f32 accumulate
  return __builtin_amdgcn_wmma_f32_16x16x32_bf16(false, a, false, b,
                                                 (short)0, c, false, false);
}

// A-fragment: this lane's row data, two 8-element contiguous K-chunks
// (lane<16: K 0..15 ; lane>=16: K 16..31 — caller offsets ptr)
__device__ __forceinline__ v16bf make_a(const __bf16* c0, const __bf16* c1) {
  v8bf a0 = *(const v8bf*)c0;
  v8bf a1 = *(const v8bf*)(c0 + 8); (void)c1;
  v16bf r;
#pragma unroll
  for (int i = 0; i < 8; ++i) { r[i] = a0[i]; r[8 + i] = a1[i]; }
  return r;
}

// --------------------------- kernel 0: f32 -> bf16 -------------------------
__global__ __launch_bounds__(256)
void cvt_f32_to_bf16(const float* __restrict__ s,
                     unsigned short* __restrict__ dst, int n) {
  __bf16* d = (__bf16*)dst;
  int i = (blockIdx.x * 256 + threadIdx.x) * 4;
  if (i + 3 < n) {
    f4v v = __builtin_nontemporal_load((const f4v*)(s + i));  // read-once
    d[i + 0] = (__bf16)v.x;
    d[i + 1] = (__bf16)v.y;
    d[i + 2] = (__bf16)v.z;
    d[i + 3] = (__bf16)v.w;
  }
}

// ------------------- kernel 1: QKV projection (+RoPE) ----------------------
// One wave per 16(M) x {n0, n0+32}(N) tile pair so RoPE partners (d, d+32)
// are both in registers. mode: 0=Q(rope), 1=K(rope) -> [bh][s][d]; 2=V -> [bh][d][s]
__global__ __launch_bounds__(32)
void qkv_proj_kernel(const unsigned short* __restrict__ Xus,
                     const unsigned short* __restrict__ Wus,
                     const float* __restrict__ bias,
                     unsigned short* __restrict__ outus, int mode) {
  const __bf16* X = (const __bf16*)Xus;
  const __bf16* W = (const __bf16*)Wus;
  __bf16* out = (__bf16*)outus;

  const int lane = threadIdx.x;
  const int col = lane & 15, hi = lane >> 4;
  const int m0 = blockIdx.x * 16;
  const int pt = blockIdx.y;            // 0..31
  const int h  = pt >> 1;
  const int dq = (pt & 1) * 16;         // quarter within head: 0 or 16
  const int n0 = h * HDIM + dq;
  const int n1 = n0 + 32;

  const __bf16* abase  = X + (size_t)(m0 + col) * DM + hi * 16;
  const __bf16* b0base = W + (size_t)(n0 + col) * DM + hi * 16;
  const __bf16* b1base = W + (size_t)(n1 + col) * DM + hi * 16;

  v8f acc0 = {};
  v8f acc1 = {};
#pragma unroll 4
  for (int k0 = 0; k0 < DM; k0 += 32) {
    __builtin_prefetch((const void*)(abase + k0 + 128), 0, 1);
    v16bf a  = make_a(abase + k0, abase + k0 + 16);
    v16bf b0 = *(const v16bf*)(b0base + k0);
    v16bf b1 = *(const v16bf*)(b1base + k0);
    acc0 = wmma_bf16(a, b0, acc0);
    acc1 = wmma_bf16(a, b1, acc1);
  }

  const float bz0 = bias[n0 + col];
  const float bz1 = bias[n1 + col];

  if (mode == 2) {
    // V: store transposed [bh][d][s] (bf16) — consecutive accumulator rows are
    // consecutive tokens => contiguous stores per lane.
#pragma unroll
    for (int r = 0; r < 8; ++r) {
      int mg = m0 + r + hi * 8;
      int b  = mg >> 11, s = mg & (SEQ - 1);
      size_t bh = (size_t)b * NH + h;
      out[(bh * HDIM + dq + col) * SEQ + s]      = (__bf16)(acc0[r] + bz0);
      out[(bh * HDIM + dq + 32 + col) * SEQ + s] = (__bf16)(acc1[r] + bz1);
    }
  } else {
    // Q/K: rotate-half RoPE, store [bh][s][d]
    const int dlo = dq + col;                           // 0..31
    const float invf = __powf(10000.f, -(float)dlo * (1.f / 32.f));
#pragma unroll
    for (int r = 0; r < 8; ++r) {
      int mg = m0 + r + hi * 8;
      int b  = mg >> 11, s = mg & (SEQ - 1);
      float sn, cs;
      __sincosf((float)s * invf, &sn, &cs);
      float xlo = acc0[r] + bz0;
      float xhi = acc1[r] + bz1;
      float olo = xlo * cs - xhi * sn;
      float ohi = xhi * cs + xlo * sn;
      size_t base = (((size_t)b * NH + h) * SEQ + s) * HDIM;
      out[base + dlo]      = (__bf16)olo;
      out[base + dlo + 32] = (__bf16)ohi;
    }
  }
}

// ----------------- kernel 2: causal softmax row stats (online) -------------
__global__ __launch_bounds__(32)
void softmax_stats_kernel(const unsigned short* __restrict__ Qus,
                          const unsigned short* __restrict__ Kus,
                          float* __restrict__ rmax, float* __restrict__ rsum) {
  const __bf16* Qh = (const __bf16*)Qus;
  const __bf16* Kh = (const __bf16*)Kus;
  const int lane = threadIdx.x, col = lane & 15, hi = lane >> 4;
  const int qb = blockIdx.x, bh = blockIdx.y;
  const int q0 = qb * 16;

  const __bf16* qrow = Qh + ((size_t)bh * SEQ + q0 + col) * HDIM + hi * 16;
  const v16bf a0 = make_a(qrow,      qrow + 16);   // d 0..31
  const v16bf a1 = make_a(qrow + 32, qrow + 48);   // d 32..63
  const __bf16* kbase = Kh + (size_t)bh * SEQ * HDIM + hi * 16;

  float m[8], ssum[8];
#pragma unroll
  for (int r = 0; r < 8; ++r) { m[r] = -3.0e38f; ssum[r] = 0.f; }

  for (int kt = 0; kt <= qb; ++kt) {
    const __bf16* krow = kbase + (size_t)(kt * 16 + col) * HDIM;
    v8f acc = {};
    acc = wmma_bf16(a0, *(const v16bf*)krow, acc);
    acc = wmma_bf16(a1, *(const v16bf*)(krow + 32), acc);
    const bool diag = (kt == qb);
    const int kcol = kt * 16 + col;
#pragma unroll
    for (int r = 0; r < 8; ++r) {
      int row = q0 + r + hi * 8;
      float v = (!diag || kcol <= row) ? acc[r] * 0.125f : -3.0e38f;
      float nm = fmaxf(m[r], v);               // per-lane online (max,sum)
      ssum[r] = ssum[r] * __expf(m[r] - nm) + __expf(v - nm);
      m[r] = nm;
    }
  }
  // merge (max,sum) across the 16 lanes holding each row
#pragma unroll
  for (int r = 0; r < 8; ++r) {
    float mm = m[r], ss = ssum[r];
#pragma unroll
    for (int off = 1; off < 16; off <<= 1) {
      float om = __shfl_xor(mm, off, 32);
      float os = __shfl_xor(ss, off, 32);
      float nm = fmaxf(mm, om);
      ss = ss * __expf(mm - nm) + os * __expf(om - nm);
      mm = nm;
    }
    m[r] = mm; ssum[r] = ss;
  }
  if (col == 0) {
#pragma unroll
    for (int r = 0; r < 8; ++r) {
      int row = q0 + r + hi * 8;
      rmax[(size_t)bh * SEQ + row] = m[r];
      rsum[(size_t)bh * SEQ + row] = ssum[r];
    }
  }
}

// --------- kernel 3: recompute scores -> write attn, fuse P@V (ctx) --------
__global__ __launch_bounds__(32)
void attn_ctx_kernel(const unsigned short* __restrict__ Qus,
                     const unsigned short* __restrict__ Kus,
                     const unsigned short* __restrict__ Vus,
                     const float* __restrict__ rmax,
                     const float* __restrict__ rsum,
                     float* __restrict__ attn,
                     unsigned short* __restrict__ Ctxus) {
  __shared__ __align__(32) __bf16 pbuf[16 * 32];   // P tile, row-major 16xK32
  const __bf16* Qh = (const __bf16*)Qus;
  const __bf16* Kh = (const __bf16*)Kus;
  const __bf16* Vt = (const __bf16*)Vus;
  __bf16* Ctx = (__bf16*)Ctxus;

  const int lane = threadIdx.x, col = lane & 15, hi = lane >> 4;
  const int qb = blockIdx.x, bh = blockIdx.y;
  const int q0 = qb * 16;

  const __bf16* qrow = Qh + ((size_t)bh * SEQ + q0 + col) * HDIM + hi * 16;
  const v16bf a0 = make_a(qrow,      qrow + 16);
  const v16bf a1 = make_a(qrow + 32, qrow + 48);
  const __bf16* kbase = Kh + (size_t)bh * SEQ * HDIM + hi * 16;

  float m[8], inv[8];
#pragma unroll
  for (int r = 0; r < 8; ++r) {
    int row = q0 + r + hi * 8;
    m[r] = rmax[(size_t)bh * SEQ + row];
    float sv = rsum[(size_t)bh * SEQ + row];
    inv[r] = (sv > 0.f) ? 1.f / sv : 0.f;
  }

  v8f cacc[4] = {};
  const int npair = (qb >> 1) + 1;               // K=32 steps for P@V
  for (int kp = 0; kp < npair; ++kp) {
#pragma unroll
    for (int sub = 0; sub < 2; ++sub) {
      const int kt = kp * 2 + sub;
      if (kt <= qb) {
        const __bf16* krow = kbase + (size_t)(kt * 16 + col) * HDIM;
        v8f acc = {};
        acc = wmma_bf16(a0, *(const v16bf*)krow, acc);
        acc = wmma_bf16(a1, *(const v16bf*)(krow + 32), acc);
        const bool diag = (kt == qb);
        const int kcol = kt * 16 + col;
#pragma unroll
        for (int r = 0; r < 8; ++r) {
          int rloc = r + hi * 8;
          int row  = q0 + rloc;
          float v = (!diag || kcol <= row) ? acc[r] * 0.125f : -3.0e38f;
          float p = __expf(v - m[r]) * inv[r];
          // streaming output: never re-read -> non-temporal, keep L2 for Q/K/V
          __builtin_nontemporal_store(
              p, &attn[((size_t)bh * SEQ + row) * SEQ + kcol]);
          pbuf[rloc * 32 + sub * 16 + col] = (__bf16)p;
        }
      } else {
#pragma unroll
        for (int r = 0; r < 8; ++r)
          pbuf[(r + hi * 8) * 32 + sub * 16 + col] = (__bf16)0.f;
      }
    }
    __syncthreads();                 // LDS C-layout -> A-layout transpose
    const __bf16* prow = pbuf + col * 32 + hi * 16;
    v16bf pa = make_a(prow, prow + 16);
    const size_t koff = (size_t)kp * 32 + hi * 16;
#pragma unroll
    for (int dt = 0; dt < 4; ++dt) {
      const __bf16* vrow = Vt + ((size_t)bh * HDIM + dt * 16 + col) * SEQ + koff;
      cacc[dt] = wmma_bf16(pa, *(const v16bf*)vrow, cacc[dt]);
    }
    __syncthreads();
  }

  // zero-fill attn above the causal diagonal (harness poisons d_out):
  // lane-strided float4 non-temporal stores (global_store_b128 nt)
  {
    const int kz0 = (qb + 1) * 16;             // first all-zero column
    const f4v z4 = {0.f, 0.f, 0.f, 0.f};
    for (int r = 0; r < 16; ++r) {
      float* rowp = attn + ((size_t)bh * SEQ + q0 + r) * SEQ;
      for (int c = kz0 + lane * 4; c < SEQ; c += 32 * 4)
        __builtin_nontemporal_store(z4, (f4v*)(rowp + c));
    }
  }

  // ctx -> bf16 [B,S,H*HD] for the output projection (re-read -> regular)
  const int b = bh >> 4, h = bh & 15;
#pragma unroll
  for (int dt = 0; dt < 4; ++dt)
#pragma unroll
    for (int r = 0; r < 8; ++r)
      Ctx[((size_t)(b * SEQ + q0 + r + hi * 8)) * DM + h * HDIM + dt * 16 + col] =
          (__bf16)cacc[dt][r];
}

// ---------------------- kernel 4: output projection ------------------------
__global__ __launch_bounds__(32)
void oproj_kernel(const unsigned short* __restrict__ Cus,
                  const unsigned short* __restrict__ Wus,
                  const float* __restrict__ bo, float* __restrict__ out) {
  const __bf16* C = (const __bf16*)Cus;
  const __bf16* W = (const __bf16*)Wus;
  const int lane = threadIdx.x, col = lane & 15, hi = lane >> 4;
  const int m0 = blockIdx.x * 16, n0 = blockIdx.y * 16;

  const __bf16* abase = C + (size_t)(m0 + col) * DM + hi * 16;
  const __bf16* bbase = W + (size_t)(n0 + col) * DM + hi * 16;

  v8f acc = {};
#pragma unroll 4
  for (int k0 = 0; k0 < DM; k0 += 32) {
    v16bf a = make_a(abase + k0, abase + k0 + 16);
    v16bf b = *(const v16bf*)(bbase + k0);
    acc = wmma_bf16(a, b, acc);
  }
  const float bz = bo[n0 + col];
#pragma unroll
  for (int r = 0; r < 8; ++r)
    __builtin_nontemporal_store(
        acc[r] + bz, &out[(size_t)(m0 + r + hi * 8) * DM + n0 + col]);
}

// ---------------------------------------------------------------------------
extern "C" void kernel_launch(void* const* d_in, const int* in_sizes, int n_in,
                              void* d_out, int out_size, void* d_ws,
                              size_t ws_size, hipStream_t stream) {
  (void)in_sizes; (void)n_in; (void)out_size; (void)ws_size;
  const float* query = (const float*)d_in[0];
  const float* keys  = (const float*)d_in[1];
  const float* vals  = (const float*)d_in[2];
  const float* Wq = (const float*)d_in[3];
  const float* bq = (const float*)d_in[4];
  const float* Wk = (const float*)d_in[5];
  const float* bk = (const float*)d_in[6];
  const float* Wv = (const float*)d_in[7];
  const float* bv = (const float*)d_in[8];
  const float* Wo = (const float*)d_in[9];
  const float* bo = (const float*)d_in[10];

  char* w = (char*)d_ws;
  size_t off = 0;
  auto carve = [&](size_t bytes) -> void* {
    void* p = w + off;
    off += (bytes + 255) & ~(size_t)255;
    return p;
  };
  const size_t actN = (size_t)NTOK * DM;   // 4,194,304
  const size_t wN   = (size_t)DM * DM;     // 1,048,576
  unsigned short* Xq  = (unsigned short*)carve(actN * 2);
  unsigned short* Xk  = (unsigned short*)carve(actN * 2);
  unsigned short* Xv  = (unsigned short*)carve(actN * 2);
  unsigned short* Wqb = (unsigned short*)carve(wN * 2);
  unsigned short* Wkb = (unsigned short*)carve(wN * 2);
  unsigned short* Wvb = (unsigned short*)carve(wN * 2);
  unsigned short* Wob = (unsigned short*)carve(wN * 2);
  unsigned short* Qh  = (unsigned short*)carve(actN * 2);   // [bh][s][d]
  unsigned short* Kh  = (unsigned short*)carve(actN * 2);   // [bh][s][d]
  unsigned short* Vt  = (unsigned short*)carve(actN * 2);   // [bh][d][s]
  unsigned short* Ctx = (unsigned short*)carve(actN * 2);   // [B,S,1024]
  float* rmax = (float*)carve((size_t)NB * NH * SEQ * 4);
  float* rsum = (float*)carve((size_t)NB * NH * SEQ * 4);

  float* out_p  = (float*)d_out;
  float* attn_p = out_p + actN;            // out first, then attn

  auto cvt = [&](const float* s, unsigned short* d, int n) {
    cvt_f32_to_bf16<<<n / 1024, 256, 0, stream>>>(s, d, n);
  };
  cvt(query, Xq, (int)actN);
  cvt(keys,  Xk, (int)actN);
  cvt(vals,  Xv, (int)actN);
  cvt(Wq, Wqb, (int)wN);
  cvt(Wk, Wkb, (int)wN);
  cvt(Wv, Wvb, (int)wN);
  cvt(Wo, Wob, (int)wN);

  qkv_proj_kernel<<<dim3(NTOK / 16, 32), 32, 0, stream>>>(Xq, Wqb, bq, Qh, 0);
  qkv_proj_kernel<<<dim3(NTOK / 16, 32), 32, 0, stream>>>(Xk, Wkb, bk, Kh, 1);
  qkv_proj_kernel<<<dim3(NTOK / 16, 32), 32, 0, stream>>>(Xv, Wvb, bv, Vt, 2);

  softmax_stats_kernel<<<dim3(SEQ / 16, NB * NH), 32, 0, stream>>>(Qh, Kh,
                                                                   rmax, rsum);
  attn_ctx_kernel<<<dim3(SEQ / 16, NB * NH), 32, 0, stream>>>(
      Qh, Kh, Vt, rmax, rsum, attn_p, Ctx);

  oproj_kernel<<<dim3(NTOK / 16, DM / 16), 32, 0, stream>>>(Ctx, Wob, bo,
                                                            out_p);
}
